// LocationAwareAttention_35253091565912
// MI455X (gfx1250) — compile-verified, matrix operand order checked by reading, not run on hardware
//
#include <hip/hip_runtime.h>

// ---------------------------------------------------------------------------
// Problem constants (reference: B=16, T=2048, D=1024)
// ---------------------------------------------------------------------------
#define BB 16
#define TT 2048
#define DD 1024
#define KTOT (2 * DD)          // concat K for q-proj + v-proj
#define MM (BB * TT)           // 32768 rows

#define TILE_M 128
#define TILE_N 128
#define TILE_K 32
#define NCHUNKS (DD / TILE_N)  // 8 partial-score slabs
#define LDSK 40                // padded LDS row stride (elements) -> 80B rows, 16B aligned

typedef __attribute__((ext_vector_type(16))) __bf16 v16bf;
typedef __attribute__((ext_vector_type(8)))  __bf16 v8bf;
typedef __attribute__((ext_vector_type(8)))  float  v8f;

union Frag16 { v16bf v; v8bf h[2]; };

__device__ __forceinline__ unsigned short f2bf_rne(float f) {
    union { float f; unsigned int u; } v; v.f = f;
    unsigned int u = v.u;
    unsigned int r = u + 0x7fffu + ((u >> 16) & 1u);   // round-to-nearest-even
    return (unsigned short)(r >> 16);
}

// ---------------------------------------------------------------------------
// Kernel 1a: query||value (fp32) -> X bf16, layout X[m][k], k in [0,2048)
// grid = MM blocks, 256 threads; each thread converts 4 q-elems + 4 v-elems
// ---------------------------------------------------------------------------
__global__ __launch_bounds__(256)
void convert_X_kernel(const float* __restrict__ query,
                      const float* __restrict__ value,
                      unsigned short* __restrict__ X) {
    const int m = blockIdx.x;
    const int k = threadIdx.x * 4;

    float4 q = *(const float4*)(query + (size_t)m * DD + k);
    float4 v = *(const float4*)(value + (size_t)m * DD + k);

    uint2 pq, pv;
    pq.x = (unsigned)f2bf_rne(q.x) | ((unsigned)f2bf_rne(q.y) << 16);
    pq.y = (unsigned)f2bf_rne(q.z) | ((unsigned)f2bf_rne(q.w) << 16);
    pv.x = (unsigned)f2bf_rne(v.x) | ((unsigned)f2bf_rne(v.y) << 16);
    pv.y = (unsigned)f2bf_rne(v.z) | ((unsigned)f2bf_rne(v.w) << 16);

    *(uint2*)(X + (size_t)m * KTOT + k)      = pq;
    *(uint2*)(X + (size_t)m * KTOT + DD + k) = pv;
}

// ---------------------------------------------------------------------------
// Kernel 1b: Wq||Wv (fp32) -> W bf16, layout W[e][k] (rows contiguous in K)
// grid = DD blocks, 256 threads
// ---------------------------------------------------------------------------
__global__ __launch_bounds__(256)
void convert_W_kernel(const float* __restrict__ Wq,
                      const float* __restrict__ Wv,
                      unsigned short* __restrict__ W) {
    const int e = blockIdx.x;
    const int k = threadIdx.x * 4;

    float4 a = *(const float4*)(Wq + (size_t)e * DD + k);
    float4 b = *(const float4*)(Wv + (size_t)e * DD + k);

    uint2 pa, pb;
    pa.x = (unsigned)f2bf_rne(a.x) | ((unsigned)f2bf_rne(a.y) << 16);
    pa.y = (unsigned)f2bf_rne(a.z) | ((unsigned)f2bf_rne(a.w) << 16);
    pb.x = (unsigned)f2bf_rne(b.x) | ((unsigned)f2bf_rne(b.y) << 16);
    pb.y = (unsigned)f2bf_rne(b.z) | ((unsigned)f2bf_rne(b.w) << 16);

    *(uint2*)(W + (size_t)e * KTOT + k)      = pa;
    *(uint2*)(W + (size_t)e * KTOT + DD + k) = pb;
}

// ---------------------------------------------------------------------------
// Kernel 2: fused (q+v) projection GEMM + conv/bias/tanh/w_score epilogue.
//   grid = (NCHUNKS, MM/TILE_M) = (8, 256), block = 256 (8 waves, wave32)
//   wave grid: 4 waves along M (32 rows each), 2 along N (64 cols each)
//   each wave: 2x4 subtiles of v_wmma_f32_16x16x32_bf16, fp32 accum
//   writes score_part[chunk][m]  (no atomics -> deterministic)
// ---------------------------------------------------------------------------
__global__ __launch_bounds__(256)
void fused_gemm_score_kernel(const unsigned short* __restrict__ X,
                             const unsigned short* __restrict__ W,
                             const float* __restrict__ last_attn,
                             const float* __restrict__ conv_w,   // (D,1,3) flat
                             const float* __restrict__ conv_b,
                             const float* __restrict__ bias,
                             const float* __restrict__ w_score,
                             float* __restrict__ score_part) {
    __shared__ __align__(16) unsigned short At[TILE_M * LDSK];
    __shared__ __align__(16) unsigned short Bt[TILE_N * LDSK];
    __shared__ float laTile[TILE_M + 2];

    const int tid   = threadIdx.x;
    const int tileN = blockIdx.x * TILE_N;
    const int tileM = blockIdx.y * TILE_M;

    const int bb = tileM >> 11;     // tileM / TT   (tiles never straddle a batch)
    const int t0 = tileM & (TT - 1);

    // stage last_attn halo: laTile[i] = la[bb, t0-1+i], zero-padded at batch edges
    if (tid < TILE_M + 2) {
        int t = t0 - 1 + tid;
        laTile[tid] = (t >= 0 && t < TT) ? last_attn[bb * TT + t] : 0.0f;
    }

    const int wid  = tid >> 5;
    const int lane = tid & 31;
    const int half = lane >> 4;
    const int l    = lane & 15;
    const int waveRow = (wid & 3) * 32;   // 0,32,64,96
    const int waveCol = (wid >> 2) * 64;  // 0,64

    v8f acc[2][4];
    const v8f vz = {0.f, 0.f, 0.f, 0.f, 0.f, 0.f, 0.f, 0.f};
#pragma unroll
    for (int s = 0; s < 2; ++s)
#pragma unroll
        for (int j = 0; j < 4; ++j) acc[s][j] = vz;

    // staging assignment: 2 threads per row, 16 bf16 (32B) each
    const int srow = tid >> 1;
    const int sh   = tid & 1;
    const unsigned short* gA = X + (size_t)(tileM + srow) * KTOT + sh * 16;
    const unsigned short* gB = W + (size_t)(tileN + srow) * KTOT + sh * 16;
    uint4* lA = (uint4*)&At[srow * LDSK + sh * 16];
    uint4* lB = (uint4*)&Bt[srow * LDSK + sh * 16];

    for (int kt = 0; kt < KTOT / TILE_K; ++kt) {
        const uint4* pa = (const uint4*)(gA + kt * TILE_K);
        const uint4* pb = (const uint4*)(gB + kt * TILE_K);
        uint4 a0 = pa[0], a1 = pa[1];
        uint4 b0 = pb[0], b1 = pb[1];

        __syncthreads();               // previous iteration's reads done
        lA[0] = a0; lA[1] = a1;
        lB[0] = b0; lB[1] = b1;
        __syncthreads();               // tile visible

        if (kt + 1 < KTOT / TILE_K) {  // CDNA5 global_prefetch_b8 for next tile
            __builtin_prefetch(gA + (kt + 1) * TILE_K, 0, 1);
            __builtin_prefetch(gB + (kt + 1) * TILE_K, 0, 1);
        }

        // A fragments: lane<16 -> M=l, K={0..7,16..23}; lane>=16 -> K={8..15,24..31}
        Frag16 fa[2], fb[4];
#pragma unroll
        for (int s = 0; s < 2; ++s) {
            const unsigned short* p = &At[(waveRow + s * 16 + l) * LDSK + half * 8];
            fa[s].h[0] = *(const v8bf*)p;
            fa[s].h[1] = *(const v8bf*)(p + 16);
        }
#pragma unroll
        for (int j = 0; j < 4; ++j) {
            const unsigned short* p = &Bt[(waveCol + j * 16 + l) * LDSK + half * 8];
            fb[j].h[0] = *(const v8bf*)p;
            fb[j].h[1] = *(const v8bf*)(p + 16);
        }

#pragma unroll
        for (int s = 0; s < 2; ++s)
#pragma unroll
            for (int j = 0; j < 4; ++j)
                acc[s][j] = __builtin_amdgcn_wmma_f32_16x16x32_bf16(
                    false, fa[s].v, false, fb[j].v, (short)0, acc[s][j],
                    false, false);
    }

    // ---- epilogue: energy = tanh(qv + conv + bias); score += energy . w_score
    float w0[4], w1[4], w2[4], cb[4], bi[4], ws[4];
#pragma unroll
    for (int j = 0; j < 4; ++j) {
        int e = tileN + waveCol + j * 16 + l;
        w0[j] = conv_w[e * 3 + 0];
        w1[j] = conv_w[e * 3 + 1];
        w2[j] = conv_w[e * 3 + 2];
        cb[j] = conv_b[e];
        bi[j] = bias[e];
        ws[j] = w_score[e];
    }

#pragma unroll
    for (int s = 0; s < 2; ++s) {
#pragma unroll
        for (int r = 0; r < 8; ++r) {
            const int mLocal = waveRow + s * 16 + half * 8 + r;
            const float lm = laTile[mLocal];
            const float lc = laTile[mLocal + 1];
            const float lp = laTile[mLocal + 2];
            float rs = 0.0f;
#pragma unroll
            for (int j = 0; j < 4; ++j) {
                float en = tanhf(acc[s][j][r] + cb[j] + bi[j] +
                                 w0[j] * lm + w1[j] * lc + w2[j] * lp);
                rs += en * ws[j];
            }
            // reduce the 16 lanes that share this row (halves stay disjoint)
            rs += __shfl_xor(rs, 1, 32);
            rs += __shfl_xor(rs, 2, 32);
            rs += __shfl_xor(rs, 4, 32);
            rs += __shfl_xor(rs, 8, 32);
            if (l == 0)
                score_part[(size_t)blockIdx.x * MM + tileM + mLocal] = rs;
        }
    }
}

// ---------------------------------------------------------------------------
// Kernel 3: score -> sigmoid -> L1-normalize per batch. grid=(B), block=256
// ---------------------------------------------------------------------------
__global__ __launch_bounds__(256)
void score_to_attn_kernel(const float* __restrict__ score_part,
                          const unsigned char* __restrict__ mask,
                          const float* __restrict__ b_score,
                          float* __restrict__ attn) {
    const int b = blockIdx.x;
    const int tid = threadIdx.x;
    const float bs = b_score[0];

    __shared__ float red[256];
    __shared__ float totalS;

    float sum = 0.0f;
    for (int t = tid; t < TT; t += 256) {
        const int m = b * TT + t;
        float s = bs;
        for (int c = 0; c < NCHUNKS; ++c)          // deterministic order
            s += score_part[(size_t)c * MM + m];
        if (mask[m]) s = -1e30f;
        float sc = 1.0f / (1.0f + expf(-s));
        attn[m] = sc;
        sum += sc;
    }
    red[tid] = sum;
    __syncthreads();
    for (int o = 128; o > 0; o >>= 1) {
        if (tid < o) red[tid] += red[tid + o];
        __syncthreads();
    }
    if (tid == 0) totalS = red[0];
    __syncthreads();
    const float inv = 1.0f / totalS;
    for (int t = tid; t < TT; t += 256) attn[b * TT + t] *= inv;
}

// ---------------------------------------------------------------------------
// Kernel 4: context[b,d] = sum_t attn[b,t] * value[b,t,d]
//   grid = (B, D/256), block = 256; attn row cached in LDS
// ---------------------------------------------------------------------------
__global__ __launch_bounds__(256)
void context_kernel(const float* __restrict__ attn,
                    const float* __restrict__ value,
                    float* __restrict__ context) {
    const int b = blockIdx.x;
    const int d = blockIdx.y * 256 + threadIdx.x;

    __shared__ float sA[TT];
    for (int t = threadIdx.x; t < TT; t += 256) sA[t] = attn[b * TT + t];
    __syncthreads();

    const float* vp = value + ((size_t)b * TT) * DD + d;
    float acc = 0.0f;
#pragma unroll 4
    for (int t = 0; t < TT; ++t) acc += sA[t] * vp[(size_t)t * DD];
    context[b * DD + d] = acc;
}

// ---------------------------------------------------------------------------
// Host launcher
// ---------------------------------------------------------------------------
extern "C" void kernel_launch(void* const* d_in, const int* in_sizes, int n_in,
                              void* d_out, int out_size, void* d_ws, size_t ws_size,
                              hipStream_t stream) {
    (void)in_sizes; (void)n_in; (void)out_size; (void)ws_size;

    const float*         query     = (const float*)d_in[0];
    const float*         value     = (const float*)d_in[1];
    const unsigned char* mask      = (const unsigned char*)d_in[2];
    const float*         last_attn = (const float*)d_in[3];
    const float*         conv_w    = (const float*)d_in[4];
    const float*         conv_b    = (const float*)d_in[5];
    const float*         Wq        = (const float*)d_in[6];
    const float*         Wv        = (const float*)d_in[7];
    const float*         w_score   = (const float*)d_in[8];
    const float*         b_score   = (const float*)d_in[9];
    const float*         bias      = (const float*)d_in[10];

    float* out     = (float*)d_out;
    float* context = out;               // (B, D)
    float* attn    = out + BB * DD;     // (B, T)

    // workspace layout: X bf16 (128 MB) | W bf16 (4 MB) | score partials (1 MB)
    unsigned short* Xbf = (unsigned short*)d_ws;
    unsigned short* Wbf = Xbf + (size_t)MM * KTOT;
    float* score_part   = (float*)(Wbf + (size_t)DD * KTOT);

    convert_X_kernel<<<MM, 256, 0, stream>>>(query, value, Xbf);
    convert_W_kernel<<<DD, 256, 0, stream>>>(Wq, Wv, Wbf);

    dim3 gGemm(NCHUNKS, MM / TILE_M);   // (8, 256)
    fused_gemm_score_kernel<<<gGemm, 256, 0, stream>>>(
        Xbf, Wbf, last_attn, conv_w, conv_b, bias, w_score, score_part);

    score_to_attn_kernel<<<BB, 256, 0, stream>>>(score_part, mask, b_score, attn);

    dim3 gCtx(BB, DD / 256);            // (16, 4)
    context_kernel<<<gCtx, 256, 0, stream>>>(attn, value, context);
}
